// SwitchAttention_1967095021978
// MI455X (gfx1250) — matrix operand order repaired and checked by reading, not compiled, not run on hardware
//
#include <hip/hip_runtime.h>
#include <hip/hip_bf16.h>

// ---------------------------------------------------------------------------
// Types for WMMA fragments (gfx1250, wave32)
// ---------------------------------------------------------------------------
typedef _Float16 v16h __attribute__((ext_vector_type(16)));
typedef _Float16 v8h  __attribute__((ext_vector_type(8)));
typedef _Float16 v4h  __attribute__((ext_vector_type(4)));
typedef float    v8f  __attribute__((ext_vector_type(8)));
typedef float    v4f  __attribute__((ext_vector_type(4)));

union AF16 { v16h v; v8h h[2]; };

// Problem constants
#define BB 2
#define SS 2048
#define DD 1024
#define HH 16
#define DKK 64
#define MM (BB * SS)      // 4096

// ---------------------------------------------------------------------------
// Fragment loaders.
//   A fragment (16x32 f16, MxK): lane L holds row m = L&15.
//     elements j=0..7  -> k = half*8 + j        (half = L>>4)
//     elements j=8..15 -> k = 16 + half*8 + (j-8)
//   B fragment (32x16 f16, KxN): lane L holds col n = L&15.
//     elements j=0..7  -> k = half*8 + j; j=8..15 -> k = 16 + half*8 + (j-8)
//   rowp points to the start of the row (A: row m; B: column n of B == row of
//   the K-contiguous source matrix).
// ---------------------------------------------------------------------------
__device__ __forceinline__ v16h load_a_frag(const _Float16* rowp, int k0) {
  const int half = (threadIdx.x >> 4) & 1;
  AF16 f;
  f.h[0] = *(const v8h*)(rowp + k0 + half * 8);
  f.h[1] = *(const v8h*)(rowp + k0 + 16 + half * 8);
  return f.v;
}

__device__ __forceinline__ v16h load_b_frag(const _Float16* rowp, int k0) {
  const int half = (threadIdx.x >> 4) & 1;
  AF16 f;
  f.h[0] = *(const v8h*)(rowp + k0 + half * 8);
  f.h[1] = *(const v8h*)(rowp + k0 + 16 + half * 8);
  return f.v;
}

__device__ __forceinline__ v8f wmma_f16(v16h a, v16h b, v8f c) {
  return __builtin_amdgcn_wmma_f32_16x16x32_f16(false, a, false, b,
                                                (short)0, c, false, false);
}

// ---------------------------------------------------------------------------
// f32 -> f16 conversion kernels (4 elements / thread)
// ---------------------------------------------------------------------------
__global__ void cvt_f32_f16(const float* __restrict__ src,
                            _Float16* __restrict__ dst, int n4) {
  int i = blockIdx.x * blockDim.x + threadIdx.x;
  if (i < n4) {
    v4f v = *(const v4f*)(src + (size_t)i * 4);
    *(v4h*)(dst + (size_t)i * 4) = __builtin_convertvector(v, v4h);
  }
}

__global__ void cvt_f32_f16_sel(const float* __restrict__ a,
                                const float* __restrict__ b,
                                const int* __restrict__ flag,
                                _Float16* __restrict__ dst, int n4) {
  const float* src = (flag[0] != 0) ? b : a;
  int i = blockIdx.x * blockDim.x + threadIdx.x;
  if (i < n4) {
    v4f v = *(const v4f*)(src + (size_t)i * 4);
    *(v4h*)(dst + (size_t)i * 4) = __builtin_convertvector(v, v4h);
  }
}

// ---------------------------------------------------------------------------
// NT GEMM: C[m,n] = sum_k A[m,k] * W[n,k]; M=4096, N=1024, K=1024.
// One wave computes a 16(M) x 64(N) tile. 8 waves / block, 512 blocks.
// mode 0/1: store f16 as [B,H,S,DK]   (Q / K)
// mode 2  : store f16 as [B,H,DK,S]   (V transposed)
// mode 3  : store f32 plain [m,n]     (final output projection)
// ---------------------------------------------------------------------------
__global__ __launch_bounds__(256) void gemm_nt(const _Float16* __restrict__ A,
                                               const _Float16* __restrict__ W,
                                               void* __restrict__ Dst,
                                               int mode) {
  const int Kd = DD;
  const int wave = blockIdx.x * 8 + (threadIdx.x >> 5);
  const int lane = threadIdx.x & 31;
  const int nlo = lane & 15;
  const int halfq = lane >> 4;

  const int mt = wave >> 4;        // 0..255
  const int ns = wave & 15;        // 0..15
  const int m0 = mt * 16;
  const int n0 = ns * 64;

  const _Float16* arow = A + (size_t)(m0 + nlo) * Kd;
  const _Float16* w0 = W + (size_t)(n0 + 0 + nlo) * Kd;
  const _Float16* w1 = W + (size_t)(n0 + 16 + nlo) * Kd;
  const _Float16* w2 = W + (size_t)(n0 + 32 + nlo) * Kd;
  const _Float16* w3 = W + (size_t)(n0 + 48 + nlo) * Kd;

  v8f c[4] = {};
  for (int k0 = 0; k0 < Kd; k0 += 32) {
    v16h a = load_a_frag(arow, k0);
    c[0] = wmma_f16(a, load_b_frag(w0, k0), c[0]);
    c[1] = wmma_f16(a, load_b_frag(w1, k0), c[1]);
    c[2] = wmma_f16(a, load_b_frag(w2, k0), c[2]);
    c[3] = wmma_f16(a, load_b_frag(w3, k0), c[3]);
  }

  // Epilogue. C layout: vgpr r -> row m0 + r + 8*half, col n0 + 16*t + nlo.
  if (mode == 3) {
    float* out = (float*)Dst;
    for (int t = 0; t < 4; ++t)
      for (int r = 0; r < 8; ++r) {
        int m = m0 + r + 8 * halfq;
        int n = n0 + t * 16 + nlo;
        out[(size_t)m * DD + n] = c[t][r];
      }
  } else {
    _Float16* out = (_Float16*)Dst;
    for (int t = 0; t < 4; ++t)
      for (int r = 0; r < 8; ++r) {
        int m = m0 + r + 8 * halfq;
        int n = n0 + t * 16 + nlo;
        int b = m >> 11, s = m & (SS - 1);
        int h = n >> 6, dk = n & (DKK - 1);
        size_t idx;
        if (mode == 2)  // V^T : [B,H,DK,S]
          idx = ((size_t)((b * HH + h) * DKK + dk)) * SS + s;
        else            // Q/K : [B,H,S,DK]
          idx = ((size_t)((b * HH + h) * SS + s)) * DKK + dk;
        out[idx] = (_Float16)c[t][r];
      }
  }
}

// ---------------------------------------------------------------------------
// Flash-style attention. One wave per 16-row Q tile of one (b,h).
// 4 waves / block; 4096 waves total -> 1024 blocks of 128 threads.
// scores = Q K^T + bias (NO 1/sqrt(d) scale, per T5), online softmax,
// ctx = P V, written f16 to ctxh [B*S, H*DK].
// ---------------------------------------------------------------------------
__global__ __launch_bounds__(128) void attn_fwd(const _Float16* __restrict__ Qh,
                                                const _Float16* __restrict__ Kh,
                                                const _Float16* __restrict__ Vth,
                                                const float* __restrict__ bias,
                                                _Float16* __restrict__ ctxh) {
  __shared__ _Float16 lds[4][16 * 40];  // per-wave 16x32 P tile, stride 40

  const int w = threadIdx.x >> 5;
  const int lane = threadIdx.x & 31;
  const int nlo = lane & 15;
  const int halfq = lane >> 4;

  const int wid = blockIdx.x * 4 + w;
  const int qt = wid & 127;
  const int h = (wid >> 7) & (HH - 1);
  const int b = wid >> 11;
  const int q0 = qt * 16;

  const _Float16* Qbase = Qh + ((size_t)(b * HH + h) * SS) * DKK;
  const _Float16* Kbase = Kh + ((size_t)(b * HH + h) * SS) * DKK;
  const _Float16* Vbase = Vth + ((size_t)(b * HH + h) * DKK) * SS;
  const float* bq = bias + ((size_t)h * SS + q0) * SS;  // bias has no batch dim

  // Q tile as two A fragments (dk 0..31, 32..63)
  const _Float16* qrow = Qbase + (size_t)(q0 + nlo) * DKK;
  const v16h aq0 = load_a_frag(qrow, 0);
  const v16h aq1 = load_a_frag(qrow, 32);

  v8f acc[4] = {};          // ctx: 16 q x 64 dk
  float mrun[8], lrun[8];   // per-row running max / sum (rows r + 8*halfq)
  for (int r = 0; r < 8; ++r) { mrun[r] = -1e30f; lrun[r] = 0.0f; }

  _Float16* myl = lds[w];

  for (int kb = 0; kb < SS; kb += 32) {
    v8f s[2] = {};
    for (int t = 0; t < 2; ++t) {
      const _Float16* krow = Kbase + (size_t)(kb + t * 16 + nlo) * DKK;
      s[t] = wmma_f16(aq0, load_b_frag(krow, 0), s[t]);
      s[t] = wmma_f16(aq1, load_b_frag(krow, 32), s[t]);
      const float* bp = bq + (size_t)kb + t * 16 + nlo;
      for (int r = 0; r < 8; ++r)
        s[t][r] += bp[(size_t)(r + 8 * halfq) * SS];
    }

    // online softmax update
    for (int r = 0; r < 8; ++r) {
      float bm = fmaxf(s[0][r], s[1][r]);
      for (int off = 1; off < 16; off <<= 1)
        bm = fmaxf(bm, __shfl_xor(bm, off, 32));
      float nm = fmaxf(mrun[r], bm);
      float sc = __expf(mrun[r] - nm);
      mrun[r] = nm;
      lrun[r] *= sc;
      acc[0][r] *= sc; acc[1][r] *= sc; acc[2][r] *= sc; acc[3][r] *= sc;
      float p0 = __expf(s[0][r] - nm);
      float p1 = __expf(s[1][r] - nm);
      s[0][r] = p0; s[1][r] = p1;
      float ls = p0 + p1;
      for (int off = 1; off < 16; off <<= 1)
        ls += __shfl_xor(ls, off, 32);
      lrun[r] += ls;
    }

    // P (C-fragment layout) -> LDS -> A-fragment layout
    for (int t = 0; t < 2; ++t)
      for (int r = 0; r < 8; ++r)
        myl[(r + 8 * halfq) * 40 + t * 16 + nlo] = (_Float16)s[t][r];
    __syncthreads();
    const v16h pa = load_a_frag(myl + nlo * 40, 0);
    __syncthreads();

    // ctx += P * V  (B fragments from V^T rows: contiguous in s)
    for (int t4 = 0; t4 < 4; ++t4) {
      const _Float16* vrow = Vbase + (size_t)(t4 * 16 + nlo) * SS + kb;
      acc[t4] = wmma_f16(pa, load_b_frag(vrow, 0), acc[t4]);
    }
  }

  // normalize and store ctx as f16 [B*S, H*DK]
  float inv[8];
  for (int r = 0; r < 8; ++r) inv[r] = 1.0f / lrun[r];
  for (int t4 = 0; t4 < 4; ++t4)
    for (int r = 0; r < 8; ++r) {
      size_t row = (size_t)(b * SS + q0 + r + 8 * halfq);
      ctxh[row * (HH * DKK) + h * DKK + t4 * 16 + nlo] =
          (_Float16)(acc[t4][r] * inv[r]);
    }
}

// ---------------------------------------------------------------------------
// Host-side launcher
// ---------------------------------------------------------------------------
extern "C" void kernel_launch(void* const* d_in, const int* in_sizes, int n_in,
                              void* d_out, int out_size, void* d_ws, size_t ws_size,
                              hipStream_t stream) {
  const float* hidden = (const float*)d_in[0];
  const float* bias   = (const float*)d_in[2];
  const float* kvs    = (const float*)d_in[3];
  const float* wq     = (const float*)d_in[4];
  const float* wk     = (const float*)d_in[5];
  const float* wv     = (const float*)d_in[6];
  const float* wo     = (const float*)d_in[7];
  const int*   isdec  = (const int*)d_in[8];

  // workspace layout (bytes), all 256-aligned
  char* ws = (char*)d_ws;
  _Float16* hiddenh = (_Float16*)(ws + 0);          //  8 MiB
  _Float16* kvh     = (_Float16*)(ws + 8388608);    //  8 MiB
  _Float16* wqh     = (_Float16*)(ws + 16777216);   //  2 MiB
  _Float16* wkh     = (_Float16*)(ws + 18874368);   //  2 MiB
  _Float16* wvh     = (_Float16*)(ws + 20971520);   //  2 MiB
  _Float16* woh     = (_Float16*)(ws + 23068672);   //  2 MiB
  _Float16* Qh      = (_Float16*)(ws + 25165824);   //  8 MiB
  _Float16* Kh      = (_Float16*)(ws + 33554432);   //  8 MiB
  _Float16* Vth     = (_Float16*)(ws + 41943040);   //  8 MiB
  _Float16* ctxh    = (_Float16*)(ws + 50331648);   //  8 MiB
  if (ws_size < 58720256) return;

  // 1) f32 -> f16 conversions
  cvt_f32_f16<<<4096, 256, 0, stream>>>(hidden, hiddenh, 1048576);
  cvt_f32_f16_sel<<<4096, 256, 0, stream>>>(hidden, kvs, isdec, kvh, 1048576);
  cvt_f32_f16<<<1024, 256, 0, stream>>>(wq, wqh, 262144);
  cvt_f32_f16<<<1024, 256, 0, stream>>>(wk, wkh, 262144);
  cvt_f32_f16<<<1024, 256, 0, stream>>>(wv, wvh, 262144);
  cvt_f32_f16<<<1024, 256, 0, stream>>>(wo, woh, 262144);

  // 2) QKV projections (WMMA GEMMs)
  gemm_nt<<<512, 256, 0, stream>>>(hiddenh, wqh, Qh, 0);
  gemm_nt<<<512, 256, 0, stream>>>(kvh, wkh, Kh, 1);
  gemm_nt<<<512, 256, 0, stream>>>(kvh, wvh, Vth, 2);

  // 3) attention with online softmax (WMMA)
  attn_fwd<<<1024, 128, 0, stream>>>(Qh, Kh, Vth, bias, ctxh);

  // 4) output projection straight into d_out (f32)
  gemm_nt<<<512, 256, 0, stream>>>(ctxh, woh, d_out, 3);

  // 5) second output: position_bias passthrough
  hipMemcpyAsync((float*)d_out + 4194304, bias, (size_t)67108864 * 4,
                 hipMemcpyDeviceToDevice, stream);
}
